// Zipformer2EncoderLayer_43722767073780
// MI455X (gfx1250) — compile-verified, matrix operand order checked by reading, not run on hardware
//
#include <hip/hip_runtime.h>

#ifndef __has_builtin
#define __has_builtin(x) 0
#endif

typedef __attribute__((ext_vector_type(16))) __bf16 v16bf;
typedef __attribute__((ext_vector_type(4)))  __bf16 v4bf;
typedef __attribute__((ext_vector_type(8)))  float  v8f;
typedef unsigned int u32x4 __attribute__((ext_vector_type(4)));
typedef int i32x4 __attribute__((ext_vector_type(4)));
typedef int i32x8 __attribute__((ext_vector_type(8)));

#define TS   2048      // sequence length S
#define TD   512       // model dim D
#define THD  8         // heads
#define TKV  2176      // KV length
#define TL   128       // cache length L
#define TPOS 4223      // POS_LEN
#define TATT 384       // ATT

__device__ __forceinline__ float dsoftplus(float x) { return (x > 20.f) ? x : log1pf(expf(x)); }
__device__ __forceinline__ float dswooshl(float x)  { return dsoftplus(x - 4.f) - 0.08f * x - 0.035f; }
__device__ __forceinline__ float dswooshr(float x)  { return dsoftplus(x - 1.f) - 0.08f * x - 0.313261687f; }

// ------------------------------------------------------------------
// Generic batched WMMA GEMM:  C = act(A @ B + bias)
// Compile-time specialized on ABF (A is bf16), TRB (B transposed),
// CBF (C stored bf16) so the K-loop has no uniform-flag branches.
// bf16 v_wmma_f32_16x16x32_bf16 compute, fp32 accumulate.
// 256 threads = 8 waves; block tile 128x128x32; per-wave 2x4 16x16 tiles.
// LDS tiles in WMMA fragment order [16-tile][lane][16 bf16] -> fragment
// load is one 32B ds read.  Software-pipelined: tile k+1 is fetched to
// registers while tile k is consumed by the matrix pipe.
// 16-bit layout: elem e of lane L holds K=(e/8)*16+(L/16)*8+(e%8), row=L&15.
// Inverse: lane=(m&15)|(((k>>3)&1)<<4), e=((k>>4)<<3)+(k&7).
// ------------------------------------------------------------------
#define BM 128
#define BN 128
#define BK 32

template <int ABF, int TRB, int CBF>
__global__ __launch_bounds__(256) void k_gemm(
    const void* __restrict__ Av, const float* __restrict__ B,
    const float* __restrict__ bias, void* __restrict__ Cv,
    int M, int N, int K, int lda, int ldb, int ldc,
    long long bsA, long long bsB, long long bsC, int act)
{
  __shared__ __bf16 sA[8][32][16];   // 8 row-tiles of BM=128 (8 KB)
  __shared__ __bf16 sB[8][32][16];   // 8 col-tiles of BN=128 (8 KB)
  const float*  Af = ABF ? nullptr : (const float*)Av + (long long)blockIdx.z * bsA;
  const __bf16* Ah = ABF ? (const __bf16*)Av + (long long)blockIdx.z * bsA : nullptr;
  B += (long long)blockIdx.z * bsB;
  const int bm = blockIdx.y * BM, bn = blockIdx.x * BN;
  const int tid = threadIdx.x, lane = tid & 31, wave = tid >> 5;
  const int wm = (wave >> 1) * 32;   // 0,32,64,96
  const int wn = (wave & 1) * 64;    // 0,64
  const bool inM = (bm + BM <= M);
  const bool inN = (bn + BN <= N);
  v8f acc[2][4] = {};

  float4 fa[4], fb[4];
  v4bf   ha[4];
  bool fastA = false, fastB = false;

  auto fetchA = [&](int k0) {
    fastA = inM && (k0 + BK <= K);
    if (fastA) {
      #pragma unroll
      for (int it = 0; it < 4; ++it) {
        int task = tid + it * 256;          // 1024 tasks: 128 rows x 8 k-groups
        int r = task >> 3, kg = (task & 7) << 2;
        if (ABF) ha[it] = *(const v4bf*)(Ah + (long long)(bm + r) * lda + (k0 + kg));
        else     fa[it] = *(const float4*)(Af + (long long)(bm + r) * lda + (k0 + kg));
      }
    } else {
      #pragma unroll
      for (int it = 0; it < 4; ++it) {
        int task = tid + it * 256;
        int r = task >> 3, kg = (task & 7) << 2;
        float4 f = {0.f, 0.f, 0.f, 0.f};
        #pragma unroll
        for (int u = 0; u < 4; ++u) {
          float v = 0.f;
          if (bm + r < M && k0 + kg + u < K) {
            long long idx = (long long)(bm + r) * lda + k0 + kg + u;
            v = ABF ? (float)Ah[idx] : Af[idx];
          }
          (&f.x)[u] = v;
        }
        fa[it] = f;
      }
    }
  };
  auto storeA = [&]() {
    #pragma unroll
    for (int it = 0; it < 4; ++it) {
      int task = tid + it * 256;
      int r = task >> 3, kg = (task & 7) << 2;
      v4bf vb;
      if (ABF && fastA) vb = ha[it];
      else { vb[0] = (__bf16)fa[it].x; vb[1] = (__bf16)fa[it].y;
             vb[2] = (__bf16)fa[it].z; vb[3] = (__bf16)fa[it].w; }
      *(v4bf*)&sA[r >> 4][(r & 15) | (((kg >> 3) & 1) << 4)][((kg >> 4) << 3) + (kg & 7)] = vb;
    }
  };
  auto fetchB = [&](int k0) {
    fastB = inN && (k0 + BK <= K);
    if (fastB) {
      #pragma unroll
      for (int it = 0; it < 4; ++it) {
        int task = tid + it * 256;          // 1024 tasks: 128 cols x 8 k-groups
        int n = task >> 3, kg = (task & 7) << 2;
        if (TRB) {
          fb[it] = *(const float4*)(B + (long long)(bn + n) * ldb + (k0 + kg));
        } else {
          const float* bp = B + (long long)(k0 + kg) * ldb + (bn + n);
          fb[it].x = bp[0]; fb[it].y = bp[ldb]; fb[it].z = bp[2 * ldb]; fb[it].w = bp[3 * ldb];
        }
      }
    } else {
      #pragma unroll
      for (int it = 0; it < 4; ++it) {
        int task = tid + it * 256;
        int n = task >> 3, kg = (task & 7) << 2;
        float4 f = {0.f, 0.f, 0.f, 0.f};
        #pragma unroll
        for (int u = 0; u < 4; ++u) {
          float v = 0.f;
          if (bn + n < N && k0 + kg + u < K)
            v = TRB ? B[(long long)(bn + n) * ldb + (k0 + kg + u)]
                    : B[(long long)(k0 + kg + u) * ldb + (bn + n)];
          (&f.x)[u] = v;
        }
        fb[it] = f;
      }
    }
  };
  auto storeB = [&]() {
    #pragma unroll
    for (int it = 0; it < 4; ++it) {
      int task = tid + it * 256;
      int n = task >> 3, kg = (task & 7) << 2;
      v4bf vb;
      vb[0] = (__bf16)fb[it].x; vb[1] = (__bf16)fb[it].y;
      vb[2] = (__bf16)fb[it].z; vb[3] = (__bf16)fb[it].w;
      *(v4bf*)&sB[n >> 4][(n & 15) | (((kg >> 3) & 1) << 4)][((kg >> 4) << 3) + (kg & 7)] = vb;
    }
  };

  fetchA(0);
  fetchB(0);
  for (int k0 = 0; k0 < K; k0 += BK) {
    storeA();
    storeB();
    __syncthreads();
    const bool last = (k0 + BK >= K);
    if (!last) {                       // loads fly under the WMMAs below
      fetchA(k0 + BK);
      fetchB(k0 + BK);
      // prefetch two tiles ahead (global_prefetch_b8)
      int prow = bm + (tid >> 1);
      if (prow < M && (k0 + 2 * BK) < K) {
        if (ABF) __builtin_prefetch(Ah + (long long)prow * lda + k0 + 2 * BK, 0, 1);
        else     __builtin_prefetch(Af + (long long)prow * lda + k0 + 2 * BK, 0, 1);
      }
    }
    const int at = wm >> 4, bt = wn >> 4;
    v16bf a0 = *(const v16bf*)&sA[at + 0][lane][0];
    v16bf a1 = *(const v16bf*)&sA[at + 1][lane][0];
    v16bf b0 = *(const v16bf*)&sB[bt + 0][lane][0];
    v16bf b1 = *(const v16bf*)&sB[bt + 1][lane][0];
    v16bf b2 = *(const v16bf*)&sB[bt + 2][lane][0];
    v16bf b3 = *(const v16bf*)&sB[bt + 3][lane][0];
    acc[0][0] = __builtin_amdgcn_wmma_f32_16x16x32_bf16(false, a0, false, b0, (short)0, acc[0][0], false, false);
    acc[0][1] = __builtin_amdgcn_wmma_f32_16x16x32_bf16(false, a0, false, b1, (short)0, acc[0][1], false, false);
    acc[0][2] = __builtin_amdgcn_wmma_f32_16x16x32_bf16(false, a0, false, b2, (short)0, acc[0][2], false, false);
    acc[0][3] = __builtin_amdgcn_wmma_f32_16x16x32_bf16(false, a0, false, b3, (short)0, acc[0][3], false, false);
    acc[1][0] = __builtin_amdgcn_wmma_f32_16x16x32_bf16(false, a1, false, b0, (short)0, acc[1][0], false, false);
    acc[1][1] = __builtin_amdgcn_wmma_f32_16x16x32_bf16(false, a1, false, b1, (short)0, acc[1][1], false, false);
    acc[1][2] = __builtin_amdgcn_wmma_f32_16x16x32_bf16(false, a1, false, b2, (short)0, acc[1][2], false, false);
    acc[1][3] = __builtin_amdgcn_wmma_f32_16x16x32_bf16(false, a1, false, b3, (short)0, acc[1][3], false, false);
    __syncthreads();
  }

  // C/D layout: VGPR r -> M = r + 8*(lane>=16), N = lane&15
  float*  Cf = CBF ? nullptr : (float*)Cv + (long long)blockIdx.z * bsC;
  __bf16* Ch = CBF ? (__bf16*)Cv + (long long)blockIdx.z * bsC : nullptr;
  const int mrow = lane & 15;
  const int rowoff = (lane >> 4) * 8;
  #pragma unroll
  for (int ti = 0; ti < 2; ++ti)
    #pragma unroll
    for (int tj = 0; tj < 4; ++tj)
      #pragma unroll
      for (int r = 0; r < 8; ++r) {
        int gm = bm + wm + ti * 16 + rowoff + r;
        int gn = bn + wn + tj * 16 + mrow;
        if (gm < M && gn < N) {
          float v = acc[ti][tj][r];
          if (bias) v += bias[gn];
          if (act == 1) v = dswooshl(v);
          else if (act == 2) v = dswooshr(v);
          if (CBF) Ch[(long long)gm * ldc + gn] = (__bf16)v;
          else     Cf[(long long)gm * ldc + gn] = v;
        }
      }
}

// ------------------------------------------------------------------
// TDM 2D tile copy: dst[r,c] = src[r,c] with independent strides.
// Each block moves 8 rows through LDS with tensor_load_to_lds /
// tensor_store_from_lds (Tensor Data Mover), synced on TENSORcnt.
// ------------------------------------------------------------------
__global__ __launch_bounds__(32) void k_tdmcopy(
    float* __restrict__ dst, int dld, const float* __restrict__ src, int sld, int cols)
{
  __shared__ float buf[8 * 384];
  const int r0 = blockIdx.x * 8;
#if __has_builtin(__builtin_amdgcn_tensor_load_to_lds) && \
    __has_builtin(__builtin_amdgcn_tensor_store_from_lds) && \
    __has_builtin(__builtin_amdgcn_s_wait_tensorcnt)
  const unsigned lds_addr = (unsigned)(unsigned long long)(void*)buf; // flat addr[31:0] == LDS byte addr
  const unsigned td0 = (unsigned)cols;
  auto mk_g1 = [&](unsigned stride) {
    i32x8 g = {};
    g[0] = (int)(2u << 16);                                    // data_size = 4 bytes
    g[1] = (int)((td0 & 0xffffu) << 16);                       // tensor_dim0 [15:0]
    g[2] = (int)(((td0 >> 16) & 0xffffu) | (8u << 16));        // tensor_dim0 hi | tensor_dim1 lo
    g[3] = (int)((td0 & 0xffffu) << 16);                       // tensor_dim1 hi=0 | tile_dim0
    g[4] = (int)8u;                                            // tile_dim1 = 8 rows
    g[5] = (int)stride;                                        // tensor_dim0_stride (elements)
    g[6] = 0; g[7] = 0;
    return g;
  };
  auto mk_g0 = [&](unsigned long long ga) {
    u32x4 g = {};
    g[0] = 1u;                                                 // count=1 valid descriptor
    g[1] = lds_addr;                                           // LDS byte address
    g[2] = (unsigned)(ga & 0xffffffffu);                       // global_addr[31:0]
    g[3] = (unsigned)((ga >> 32) & 0x1ffffffu) | (2u << 30);   // global_addr[56:32] | type=2
    return g;
  };
  const i32x4 z4 = {};
  const i32x8 z8 = {};
  (void)z8;
  u32x4 g0l = mk_g0((unsigned long long)(const void*)(src + (long long)r0 * sld));
  i32x8 g1l = mk_g1((unsigned)sld);
  u32x4 g0s = mk_g0((unsigned long long)(void*)(dst + (long long)r0 * dld));
  i32x8 g1s = mk_g1((unsigned)dld);
#if __clang_major__ >= 23
  __builtin_amdgcn_tensor_load_to_lds(g0l, g1l, z4, z4, z8, 0);
  __builtin_amdgcn_s_wait_tensorcnt(0);
  __builtin_amdgcn_tensor_store_from_lds(g0s, g1s, z4, z4, z8, 0);
#else
  __builtin_amdgcn_tensor_load_to_lds(g0l, g1l, z4, z4, 0);
  __builtin_amdgcn_s_wait_tensorcnt(0);
  __builtin_amdgcn_tensor_store_from_lds(g0s, g1s, z4, z4, 0);
#endif
  __builtin_amdgcn_s_wait_tensorcnt(0);
#else
  (void)buf;
  for (int i = threadIdx.x; i < 8 * cols; i += 32) {
    int r = r0 + i / cols, c = i % cols;
    dst[(long long)r * dld + c] = src[(long long)r * sld + c];
  }
#endif
}

// ------------------------------------------------------------------
// Element-wise / reduction kernels (row-indexed grids: no int div)
// ------------------------------------------------------------------
__global__ __launch_bounds__(256) void k_copy2d(
    float* __restrict__ dst, int dld, const float* __restrict__ src, int sld, int cols)
{
  const long long r = blockIdx.x;
  for (int c = threadIdx.x; c < cols; c += 256)
    dst[r * dld + c] = src[r * sld + c];
}

__global__ __launch_bounds__(256) void k_add(
    float* __restrict__ dst, const float* __restrict__ a, const float* __restrict__ b, int n)
{
  for (int i = blockIdx.x * 256 + threadIdx.x; i < n; i += gridDim.x * 256)
    dst[i] = a[i] + b[i];
}

// scores = scores*QHD^-0.5 + dot4(ph[s], pe[s-t+KV-1]); key padding mask
__global__ __launch_bounds__(256) void k_scorepos(
    float* __restrict__ sc, const float* __restrict__ attn_x,
    const float* __restrict__ pe, const unsigned char* __restrict__ mask)
{
  const int t = blockIdx.x * 256 + threadIdx.x;
  if (t >= TKV) return;
  const int s = blockIdx.y, h = blockIdx.z;
  const long long i = ((long long)h * TS + s) * TKV + t;
  float v = sc[i] * 0.17677669529663687f; // 32^-0.5
  const int id = s - t + TKV - 1;
  const float* ph = attn_x + (long long)s * 544 + 512 + h * 4;
  const float* pv = pe + (long long)id * 32 + h * 4;
  v += ph[0] * pv[0] + ph[1] * pv[1] + ph[2] * pv[2] + ph[3] * pv[3];
  if (t >= TL && mask[t - TL]) v = -1e30f;
  sc[i] = v;
}

// row softmax over fp32 scores -> bf16 probabilities (halves downstream traffic)
__global__ __launch_bounds__(256) void k_softmax(
    float* __restrict__ sc, __bf16* __restrict__ pr, int cols)
{
  __shared__ float red[256];
  float* row = sc + (long long)blockIdx.x * cols;
  __bf16* prow = pr + (long long)blockIdx.x * cols;
  const int tid = threadIdx.x;
  float m = -1e30f;
  for (int i = tid; i < cols; i += 256) m = fmaxf(m, row[i]);
  red[tid] = m; __syncthreads();
  for (int s = 128; s > 0; s >>= 1) { if (tid < s) red[tid] = fmaxf(red[tid], red[tid + s]); __syncthreads(); }
  m = red[0]; __syncthreads();
  float sum = 0.f;
  for (int i = tid; i < cols; i += 256) { float e = expf(row[i] - m); row[i] = e; sum += e; }
  red[tid] = sum; __syncthreads();
  for (int s = 128; s > 0; s >>= 1) { if (tid < s) red[tid] += red[tid + s]; __syncthreads(); }
  const float inv = 1.f / red[0];
  for (int i = tid; i < cols; i += 256) prow[i] = (__bf16)(row[i] * inv);
}

// dst[s,j] = x[s,384+j] * tanh(x[s,j])   (x is S x 1152)
__global__ __launch_bounds__(256) void k_tanhgate(
    float* __restrict__ dst, const float* __restrict__ x)
{
  const long long s = blockIdx.x;
  for (int j = threadIdx.x; j < TATT; j += 256)
    dst[s * TATT + j] = x[s * 1152 + TATT + j] * tanhf(x[s * 1152 + j]);
}

// dst[r,c] *= src[r,c] (strided views)
__global__ __launch_bounds__(256) void k_mulstr(
    float* __restrict__ dst, int dld, const float* __restrict__ src, int sld, int cols)
{
  const long long r = blockIdx.x;
  for (int c = threadIdx.x; c < cols; c += 256)
    dst[r * dld + c] *= src[r * sld + c];
}

// GLU: dst[s,d] = x[s,d] * sigmoid(x[s,D+d]); zero where mask[s]
__global__ __launch_bounds__(256) void k_glu(
    float* __restrict__ dst, const float* __restrict__ x, const unsigned char* __restrict__ mask)
{
  const long long s = blockIdx.x;
  const bool msk = mask[s] != 0;
  for (int d = threadIdx.x; d < TD; d += 256) {
    float a = x[s * 2 * TD + d];
    float g = x[s * 2 * TD + TD + d];
    float v = a * (1.f / (1.f + expf(-g)));
    dst[s * TD + d] = msk ? 0.f : v;
  }
}

// conv cache extraction: dst[d,j] = g[2010+j, d], j<30
__global__ __launch_bounds__(32) void k_convcache(
    float* __restrict__ dst, const float* __restrict__ g)
{
  const int d = blockIdx.x;
  const int j = threadIdx.x;
  if (j < 30) dst[d * 30 + j] = g[(2010 + j) * TD + d];
}

// depthwise conv K=31 with (K-1) cache + swoosh_r
__global__ __launch_bounds__(256) void k_dwconv(
    float* __restrict__ out, const float* __restrict__ g,
    const float* __restrict__ cache, const float* __restrict__ wgt,
    const float* __restrict__ b)
{
  const int d = blockIdx.x * 256 + threadIdx.x;
  const int s = blockIdx.y;
  if (d >= TD) return;
  float acc = b[d];
  #pragma unroll
  for (int k = 0; k < 31; ++k) {
    int t = s + k - 30;
    float x = (t < 0) ? cache[d * 30 + (s + k)] : g[(long long)t * TD + d];
    acc += wgt[d * 31 + k] * x;
  }
  out[(long long)s * TD + d] = dswooshr(acc);
}

// bias-norm: row *= rsqrt(mean((row-bias)^2)) * exp(log_scale)
__global__ __launch_bounds__(256) void k_biasnorm(
    float* __restrict__ x, const float* __restrict__ bias, const float* __restrict__ lsp)
{
  __shared__ float red[256];
  float* row = x + (long long)blockIdx.x * TD;
  const int tid = threadIdx.x;
  float sum = 0.f;
  for (int i = tid; i < TD; i += 256) { float d = row[i] - bias[i]; sum += d * d; }
  red[tid] = sum; __syncthreads();
  for (int s = 128; s > 0; s >>= 1) { if (tid < s) red[tid] += red[tid + s]; __syncthreads(); }
  const float scale = rsqrtf(red[0] / (float)TD) * expf(lsp[0]);
  __syncthreads();
  for (int i = tid; i < TD; i += 256) row[i] *= scale;
}

// dst[s,:] = orig + (cur - orig) * scale[d]
__global__ __launch_bounds__(256) void k_bypass(
    float* __restrict__ dst, const float* __restrict__ cur,
    const float* __restrict__ orig, const float* __restrict__ scale)
{
  const int d = blockIdx.x * 256 + threadIdx.x;
  const long long i = (long long)blockIdx.y * TD + d;
  if (d < TD) dst[i] = orig[i] + (cur[i] - orig[i]) * scale[d];
}

// ------------------------------------------------------------------
// Host orchestration
// ------------------------------------------------------------------
extern "C" void kernel_launch(void* const* d_in, const int* in_sizes, int n_in,
                              void* d_out, int out_size, void* d_ws, size_t ws_size,
                              hipStream_t stream)
{
  (void)in_sizes; (void)n_in; (void)out_size; (void)ws_size;
  const float* src        = (const float*)d_in[0];
  const float* pos_emb    = (const float*)d_in[1];
  const float* cached_key = (const float*)d_in[2];
  const float* cached_nl  = (const float*)d_in[3];
  const float* cached_v1  = (const float*)d_in[4];
  const float* cached_v2  = (const float*)d_in[5];
  const float* cached_c1  = (const float*)d_in[6];
  const float* cached_c2  = (const float*)d_in[7];
  const unsigned char* mask = (const unsigned char*)d_in[8];
  const float* attn_in_w  = (const float*)d_in[9];
  const float* attn_in_b  = (const float*)d_in[10];
  const float* attn_pos_w = (const float*)d_in[11];
  const float* ff1_w1 = (const float*)d_in[12]; const float* ff1_b1 = (const float*)d_in[13];
  const float* ff1_w2 = (const float*)d_in[14]; const float* ff1_b2 = (const float*)d_in[15];
  const float* ff2_w1 = (const float*)d_in[16]; const float* ff2_b1 = (const float*)d_in[17];
  const float* ff2_w2 = (const float*)d_in[18]; const float* ff2_b2 = (const float*)d_in[19];
  const float* ff3_w1 = (const float*)d_in[20]; const float* ff3_b1 = (const float*)d_in[21];
  const float* ff3_w2 = (const float*)d_in[22]; const float* ff3_b2 = (const float*)d_in[23];
  const float* na_in_w  = (const float*)d_in[24]; const float* na_in_b  = (const float*)d_in[25];
  const float* na_out_w = (const float*)d_in[26]; const float* na_out_b = (const float*)d_in[27];
  const float* sa1_in_w  = (const float*)d_in[28]; const float* sa1_in_b  = (const float*)d_in[29];
  const float* sa1_out_w = (const float*)d_in[30]; const float* sa1_out_b = (const float*)d_in[31];
  const float* sa2_in_w  = (const float*)d_in[32]; const float* sa2_in_b  = (const float*)d_in[33];
  const float* sa2_out_w = (const float*)d_in[34]; const float* sa2_out_b = (const float*)d_in[35];
  const float* c1_in_w  = (const float*)d_in[36]; const float* c1_in_b  = (const float*)d_in[37];
  const float* c1_dw_w  = (const float*)d_in[38]; const float* c1_dw_b  = (const float*)d_in[39];
  const float* c1_out_w = (const float*)d_in[40]; const float* c1_out_b = (const float*)d_in[41];
  const float* c2_in_w  = (const float*)d_in[42]; const float* c2_in_b  = (const float*)d_in[43];
  const float* c2_dw_w  = (const float*)d_in[44]; const float* c2_dw_b  = (const float*)d_in[45];
  const float* c2_out_w = (const float*)d_in[46]; const float* c2_out_b = (const float*)d_in[47];
  const float* norm_bias = (const float*)d_in[48];
  const float* norm_ls   = (const float*)d_in[49];
  const float* byp_mid   = (const float*)d_in[50];
  const float* byp_fin   = (const float*)d_in[51];

  // output layout (flat, return-tuple order)
  float* out   = (float*)d_out;
  float* o_src = out;
  float* o_key = o_src + (size_t)TS * TD;
  float* o_nl  = o_key + (size_t)TL * 256;
  float* o_v1  = o_nl  + (size_t)TL * TATT;
  float* o_v2  = o_v1  + (size_t)TL * 96;
  float* o_c1  = o_v2  + (size_t)TL * 96;
  float* o_c2  = o_c1  + (size_t)TD * 30;

  // workspace arena (all offsets stay 16B-aligned: sizes are multiples of 4)
  float* w = (float*)d_ws;
  size_t off = 0;
  auto walloc = [&](size_t n) { float* p = w + off; off += (n + 3) & ~(size_t)3; return p; };
  float*  attn_x = walloc((size_t)TS * 544);
  float*  kfull  = walloc((size_t)TKV * 256);
  float*  pe     = walloc((size_t)TPOS * 32);
  float*  scores = walloc((size_t)THD * TS * TKV);
  __bf16* probs  = (__bf16*)walloc((size_t)THD * TS * TKV / 2);   // bf16 attention weights
  float*  cur    = walloc((size_t)TS * TD);
  float*  wide   = walloc((size_t)TS * 2560);                      // fp32 view
  __bf16* wideh  = (__bf16*)wide;                                  // bf16 view (ff hidden)
  float*  tmpd   = walloc((size_t)TS * TD);
  float*  xg     = walloc((size_t)TS * TATT);
  float*  nlfull = walloc((size_t)TKV * TATT);
  float*  t384   = walloc((size_t)TS * TATT);
  float*  vx     = walloc((size_t)TS * 96);
  float*  vfull  = walloc((size_t)TKV * 96);
  float*  obuf   = walloc((size_t)TS * 96);
  float*  gbuf   = walloc((size_t)TS * TD);
  float*  cvbuf  = walloc((size_t)TS * TD);

  // static dispatch over (abf, transB, cbf) -> specialized instantiations
  auto gemm = [&](const void* A, int lda, long long bsA, int abf,
                  const float* B, int ldb, long long bsB, int transB,
                  const float* bias, void* C, int ldc, long long bsC, int cbf,
                  int M, int N, int K, int act, int batch) {
    dim3 g((N + BN - 1) / BN, (M + BM - 1) / BM, batch);
    if (abf)
      k_gemm<1, 0, 0><<<g, 256, 0, stream>>>(A, B, bias, C, M, N, K, lda, ldb, ldc, bsA, bsB, bsC, act);
    else if (transB)
      k_gemm<0, 1, 0><<<g, 256, 0, stream>>>(A, B, bias, C, M, N, K, lda, ldb, ldc, bsA, bsB, bsC, act);
    else if (cbf)
      k_gemm<0, 0, 1><<<g, 256, 0, stream>>>(A, B, bias, C, M, N, K, lda, ldb, ldc, bsA, bsB, bsC, act);
    else
      k_gemm<0, 0, 0><<<g, 256, 0, stream>>>(A, B, bias, C, M, N, K, lda, ldb, ldc, bsA, bsB, bsC, act);
  };
  // 2D strided copy; TDM path for 8-row-aligned tiles that fit staging LDS
  auto tcopy = [&](float* dst, int dld, const float* s2, int sld, int rows, int cols) {
    if ((rows % 8) == 0 && cols <= 384)
      k_tdmcopy<<<rows / 8, 32, 0, stream>>>(dst, dld, s2, sld, cols);
    else
      k_copy2d<<<rows, 256, 0, stream>>>(dst, dld, s2, sld, cols);
  };
  auto addv = [&](float* dst, const float* a, const float* b, int n) {
    k_add<<<(n + 255) / 256, 256, 0, stream>>>(dst, a, b, n);
  };

  // ---- attention weights ----
  gemm(src, TD, 0, 0, attn_in_w, 544, 0, 0, attn_in_b, attn_x, 544, 0, 0, TS, 544, TD, 0, 1);
  tcopy(kfull, 256, cached_key, 256, TL, 256);
  tcopy(kfull + TL * 256, 256, attn_x + 256, 544, TS, 256);
  gemm(pos_emb, 48, 0, 0, attn_pos_w, 32, 0, 0, nullptr, pe, 32, 0, 0, TPOS, 32, 48, 0, 1);
  // per-head scores = q @ k^T (batched over heads, transposed B)
  gemm(attn_x, 544, 32, 0, kfull, 256, 32, 1, nullptr,
       scores, TKV, (long long)TS * TKV, 0, TS, TKV, 32, 0, THD);
  k_scorepos<<<dim3((TKV + 255) / 256, TS, THD), 256, 0, stream>>>(scores, attn_x, pe, mask);
  k_softmax<<<THD * TS, 256, 0, stream>>>(scores, probs, TKV);
  tcopy(o_key, 256, attn_x + 1912 * 544 + 256, 544, TL, 256);

  // ---- ff1 (hidden activations flow bf16 producer->consumer) ----
  gemm(src, TD, 0, 0, ff1_w1, 1536, 0, 0, ff1_b1, wideh, 1536, 0, 1, TS, 1536, TD, 1, 1);
  gemm(wideh, 1536, 0, 1, ff1_w2, TD, 0, 0, ff1_b2, tmpd, TD, 0, 0, TS, TD, 1536, 0, 1);
  addv(cur, src, tmpd, TS * TD);

  // ---- nonlin attention ----
  gemm(cur, TD, 0, 0, na_in_w, 1152, 0, 0, na_in_b, wide, 1152, 0, 0, TS, 1152, TD, 0, 1);
  k_tanhgate<<<TS, 256, 0, stream>>>(xg, wide);
  tcopy(nlfull, TATT, cached_nl, TATT, TL, TATT);
  tcopy(nlfull + TL * TATT, TATT, xg, TATT, TS, TATT);
  gemm(probs, TKV, 0, 1, nlfull, TATT, 0, 0, nullptr, t384, TATT, 0, 0, TS, TATT, TKV, 0, 1);
  k_mulstr<<<TS, 256, 0, stream>>>(t384, TATT, wide + 768, 1152, TATT);
  gemm(t384, TATT, 0, 0, na_out_w, TD, 0, 0, na_out_b, tmpd, TD, 0, 0, TS, TD, TATT, 0, 1);
  addv(cur, cur, tmpd, TS * TD);
  tcopy(o_nl, TATT, xg + 1912 * TATT, TATT, TL, TATT);

  // ---- self-attention block (sa1 / sa2) ----
  auto self_attn = [&](const float* cached_v, const float* in_w, const float* in_b,
                       const float* out_w, const float* out_b, float* o_cache) {
    gemm(cur, TD, 0, 0, in_w, 96, 0, 0, in_b, vx, 96, 0, 0, TS, 96, TD, 0, 1);
    tcopy(vfull, 96, cached_v, 96, TL, 96);
    tcopy(vfull + TL * 96, 96, vx, 96, TS, 96);
    gemm(probs, TKV, (long long)TS * TKV, 1, vfull, 96, 12, 0, nullptr,
         obuf, 96, 12, 0, TS, 12, TKV, 0, THD);
    gemm(obuf, 96, 0, 0, out_w, TD, 0, 0, out_b, tmpd, TD, 0, 0, TS, TD, 96, 0, 1);
    addv(cur, cur, tmpd, TS * TD);
    tcopy(o_cache, 96, vx + 1912 * 96, 96, TL, 96);
  };

  // ---- conv block (conv1 / conv2) ----
  auto conv_blk = [&](const float* cache, const float* in_w, const float* in_b,
                      const float* dw_w, const float* dw_b,
                      const float* out_w, const float* out_b, float* o_cache) {
    gemm(cur, TD, 0, 0, in_w, 2 * TD, 0, 0, in_b, wide, 2 * TD, 0, 0, TS, 2 * TD, TD, 0, 1);
    k_glu<<<TS, 256, 0, stream>>>(gbuf, wide, mask);
    k_convcache<<<TD, 32, 0, stream>>>(o_cache, gbuf);
    k_dwconv<<<dim3(2, TS), 256, 0, stream>>>(cvbuf, gbuf, cache, dw_w, dw_b);
    gemm(cvbuf, TD, 0, 0, out_w, TD, 0, 0, out_b, tmpd, TD, 0, 0, TS, TD, TD, 0, 1);
    addv(cur, cur, tmpd, TS * TD);
  };

  self_attn(cached_v1, sa1_in_w, sa1_in_b, sa1_out_w, sa1_out_b, o_v1);
  conv_blk(cached_c1, c1_in_w, c1_in_b, c1_dw_w, c1_dw_b, c1_out_w, c1_out_b, o_c1);

  // ---- ff2 ----
  gemm(cur, TD, 0, 0, ff2_w1, 2048, 0, 0, ff2_b1, wideh, 2048, 0, 1, TS, 2048, TD, 1, 1);
  gemm(wideh, 2048, 0, 1, ff2_w2, TD, 0, 0, ff2_b2, tmpd, TD, 0, 0, TS, TD, 2048, 0, 1);
  addv(cur, cur, tmpd, TS * TD);

  // ---- mid bypass ----
  k_bypass<<<dim3(2, TS), 256, 0, stream>>>(cur, cur, src, byp_mid);

  self_attn(cached_v2, sa2_in_w, sa2_in_b, sa2_out_w, sa2_out_b, o_v2);
  conv_blk(cached_c2, c2_in_w, c2_in_b, c2_dw_w, c2_dw_b, c2_out_w, c2_out_b, o_c2);

  // ---- ff3 ----
  gemm(cur, TD, 0, 0, ff3_w1, 2560, 0, 0, ff3_b1, wideh, 2560, 0, 1, TS, 2560, TD, 1, 1);
  gemm(wideh, 2560, 0, 1, ff3_w2, TD, 0, 0, ff3_b2, tmpd, TD, 0, 0, TS, TD, 2560, 0, 1);
  addv(cur, cur, tmpd, TS * TD);

  // ---- bias norm + final bypass (writes output) ----
  k_biasnorm<<<TS, 256, 0, stream>>>(cur, norm_bias, norm_ls);
  k_bypass<<<dim3(2, TS), 256, 0, stream>>>(o_src, cur, src, byp_fin);
}
